// Experts_2594160247624
// MI455X (gfx1250) — compile-verified, matrix operand order checked by reading, not run on hardware
//
#include <hip/hip_runtime.h>
#include <hip/hip_bf16.h>
#include <cstdint>
#include <cstddef>

// Problem constants (match reference): T, D, F, E, C
#define T_TOK 8192
#define DD    2048
#define FF    8192
#define EE    8
#define CC    2048

typedef __attribute__((ext_vector_type(16))) __bf16 v16bf;
typedef __attribute__((ext_vector_type(8)))  float  v8f;
typedef int v4i_ __attribute__((vector_size(16)));   // matches builtin param type

#define AS1 __attribute__((address_space(1)))
#define AS3 __attribute__((address_space(3)))

#if __has_builtin(__builtin_amdgcn_global_load_async_to_lds_b128) && \
    __has_builtin(__builtin_amdgcn_s_wait_asynccnt)
#define USE_ASYNC_LDS 1
#else
#define USE_ASYNC_LDS 0
#endif

union Frag  { v16bf v; float4 f4[2]; };
union F8    { float4 v[2]; float f[8]; };

__device__ __forceinline__ unsigned pack_bf16(float lo, float hi) {
  unsigned short a = __builtin_bit_cast(unsigned short, (__bf16)lo);
  unsigned short b = __builtin_bit_cast(unsigned short, (__bf16)hi);
  return ((unsigned)b << 16) | (unsigned)a;
}

// branchless tanh-gelu: tanh(u) = 1 - 2/(exp(2u)+1)
__device__ __forceinline__ float gelu_tanh(float x) {
  const float k0 = 0.7978845608028654f;  // sqrt(2/pi)
  const float k1 = 0.044715f;
  float u = k0 * (x + k1 * x * x * x);
  float t = 1.0f - 2.0f * __builtin_amdgcn_rcpf(__expf(2.0f * u) + 1.0f);
  return 0.5f * x * (1.0f + t);
}

// XOR-swizzled B fragment load: 16B chunk index ^ ((col>>3)&7)
__device__ __forceinline__ float4 ld_bfrag(const __bf16* rowbase, int col, int elemoff) {
  int chunk = (elemoff >> 3) ^ ((col >> 3) & 7);
  return *(const float4*)((const char*)rowbase + chunk * 16);
}

// async (or fallback) 16B global->LDS copy
__device__ __forceinline__ void cp_b128_to_lds(const __bf16* gsrc, __bf16* ldst) {
#if USE_ASYNC_LDS
  __builtin_amdgcn_global_load_async_to_lds_b128(
      (AS1 v4i_*)gsrc, (AS3 v4i_*)ldst, 0, 0);
#else
  *(float4*)ldst = *(const float4*)gsrc;
#endif
}

__global__ void zero_out_kernel(float* __restrict__ out, int n4) {
  int i = blockIdx.x * blockDim.x + threadIdx.x;
  if (i < n4) ((float4*)out)[i] = make_float4(0.f, 0.f, 0.f, 0.f);
}

// tok[c, :] = bf16(inputs[top_idx[c, e], :]) -- vectorized f32x8 -> bf16x8
__global__ void gather_tokens_bf16(const float* __restrict__ inputs,
                                   const int* __restrict__ top_idx,
                                   int e, __bf16* __restrict__ tok) {
  int i  = blockIdx.x * blockDim.x + threadIdx.x;  // [0, C*D/8)
  int c  = i / (DD / 8);
  int d0 = (i % (DD / 8)) * 8;
  int t  = top_idx[c * EE + e];
  const float* src = inputs + (size_t)t * DD + d0;
  F8 x;
  x.v[0] = *(const float4*)(src);
  x.v[1] = *(const float4*)(src + 4);
  uint4 p;
  p.x = pack_bf16(x.f[0], x.f[1]);
  p.y = pack_bf16(x.f[2], x.f[3]);
  p.z = pack_bf16(x.f[4], x.f[5]);
  p.w = pack_bf16(x.f[6], x.f[7]);
  *(uint4*)(tok + (size_t)c * DD + d0) = p;
}

// ---------------- shared GEMM body (128x128 block, K-step 64) ----------------
// A: [M, K] bf16 row-major.  W: [K, N] fp32 row-major (expert slice).
// 8 waves: waveM = wave>>1 (4), waveN = wave&1 (2); wave tile = 32(M) x 64(N).

#define GEMM_PROLOGUE()                                                        \
  __shared__ __align__(16) __bf16 Alds[128][72];                               \
  __shared__ __align__(16) __bf16 Blds[128][72];                               \
  const int tid   = threadIdx.x;                                               \
  const int lane  = tid & 31;                                                  \
  const int wave  = tid >> 5;                                                  \
  const int waveM = wave >> 1;      /* 0..3 */                                 \
  const int waveN = wave & 1;       /* 0..1 */                                 \
  const int lr    = lane & 15;                                                 \
  const int lh    = lane >> 4;                                                 \
  const int koff  = lh * 8;                                                    \
  const int bn = blockIdx.x * 128;                                             \
  const int bm = blockIdx.y * 128;                                             \
  v8f acc[2][4] = {};                                                          \
  /* A staging: 4 chunks of 8 bf16 per thread */                               \
  const int ach = tid * 4;                                                     \
  /* B staging: 2 k-pair x 8 n per thread */                                  \
  const int bn0 = (tid & 15) * 8;                                              \
  const int bkp = tid >> 4;         /* 0..15 */

#define GEMM_STAGE(Aptr, lda, Wptr, ldb)                                       \
  {                                                                            \
    _Pragma("unroll")                                                          \
    for (int u = 0; u < 4; ++u) {                                              \
      int ch = ach + u;                                                        \
      int r = ch >> 3, c0 = (ch & 7) * 8;                                      \
      cp_b128_to_lds((Aptr) + (size_t)(bm + r) * (lda) + kb + c0,              \
                     &Alds[r][c0]);                                            \
    }                                                                          \
    _Pragma("unroll")                                                          \
    for (int u = 0; u < 2; ++u) {                                              \
      int kp = bkp + u * 16;              /* k-pair 0..31 */                   \
      const float* s0 = (Wptr) + (size_t)(kb + 2 * kp) * (ldb) + bn + bn0;     \
      const float* s1 = s0 + (ldb);                                            \
      F8 lo, hi;                                                               \
      lo.v[0] = *(const float4*)(s0);  lo.v[1] = *(const float4*)(s0 + 4);     \
      hi.v[0] = *(const float4*)(s1);  hi.v[1] = *(const float4*)(s1 + 4);     \
      _Pragma("unroll")                                                        \
      for (int j = 0; j < 8; ++j) {                                            \
        int n = bn0 + j;                                                       \
        int w = 4 * ((kp >> 2) ^ ((n >> 3) & 7)) + (kp & 3);                   \
        ((unsigned*)&Blds[n][0])[w] = pack_bf16(lo.f[j], hi.f[j]);             \
      }                                                                        \
    }                                                                          \
  }

#define GEMM_WAIT_ASYNC()                                                      \
  do { if (USE_ASYNC_LDS) __builtin_amdgcn_s_wait_asynccnt(0); } while (0)

#define GEMM_COMPUTE()                                                         \
  _Pragma("unroll")                                                            \
  for (int kk = 0; kk < 64; kk += 32) {                                        \
    Frag af[2], bfr[4];                                                        \
    _Pragma("unroll")                                                          \
    for (int tm = 0; tm < 2; ++tm) {                                           \
      const int row = waveM * 32 + tm * 16 + lr;                               \
      af[tm].f4[0] = *(const float4*)&Alds[row][kk + koff];                    \
      af[tm].f4[1] = *(const float4*)&Alds[row][kk + 16 + koff];               \
    }                                                                          \
    _Pragma("unroll")                                                          \
    for (int tn = 0; tn < 4; ++tn) {                                           \
      const int col = waveN * 64 + tn * 16 + lr;                               \
      bfr[tn].f4[0] = ld_bfrag(&Blds[col][0], col, kk + koff);                 \
      bfr[tn].f4[1] = ld_bfrag(&Blds[col][0], col, kk + 16 + koff);            \
    }                                                                          \
    _Pragma("unroll")                                                          \
    for (int tm = 0; tm < 2; ++tm)                                             \
      _Pragma("unroll")                                                        \
      for (int tn = 0; tn < 4; ++tn)                                           \
        acc[tm][tn] = __builtin_amdgcn_wmma_f32_16x16x32_bf16(                 \
            false, af[tm].v, false, bfr[tn].v, (short)0, acc[tm][tn],          \
            false, false);                                                     \
  }

// H[C,F] = bf16(gelu(tok[C,D] @ W[D,F] + bias[F]))
__global__ __launch_bounds__(256)
void gemm1_gelu_kernel(const __bf16* __restrict__ A,
                       const float*  __restrict__ W,
                       const float*  __restrict__ bias,
                       __bf16* __restrict__ H) {
  GEMM_PROLOGUE();
  for (int kb = 0; kb < DD; kb += 64) {
    GEMM_STAGE(A, DD, W, FF);
    GEMM_WAIT_ASYNC();
    __syncthreads();
    GEMM_COMPUTE();
    __syncthreads();
  }
#pragma unroll
  for (int tn = 0; tn < 4; ++tn) {
    const int col = bn + waveN * 64 + tn * 16 + lr;
    const float b = bias[col];
#pragma unroll
    for (int tm = 0; tm < 2; ++tm) {
#pragma unroll
      for (int v = 0; v < 8; ++v) {
        const int row = bm + waveM * 32 + tm * 16 + lh * 8 + v;
        H[(size_t)row * FF + col] = (__bf16)gelu_tanh(acc[tm][tn][v] + b);
      }
    }
  }
}

// out[top_idx[c,e], :] += (H[C,F] @ W[F,D] + b2[D]) * gate[top_idx[c,e], e]
__global__ __launch_bounds__(256)
void gemm2_scatter_kernel(const __bf16* __restrict__ H,
                          const float*  __restrict__ W,
                          const float*  __restrict__ bias,
                          const int*    __restrict__ top_idx,
                          const float*  __restrict__ gate,
                          int e,
                          float* __restrict__ out) {
  GEMM_PROLOGUE();
  for (int kb = 0; kb < FF; kb += 64) {
    GEMM_STAGE(H, FF, W, DD);
    GEMM_WAIT_ASYNC();
    __syncthreads();
    GEMM_COMPUTE();
    __syncthreads();
  }
#pragma unroll
  for (int tm = 0; tm < 2; ++tm) {
#pragma unroll
    for (int v = 0; v < 8; ++v) {
      const int c = bm + waveM * 32 + tm * 16 + lh * 8 + v;
      const int t = top_idx[c * EE + e];
      const float wgt = gate[(size_t)t * EE + e];
      float* orow = out + (size_t)t * DD;
#pragma unroll
      for (int tn = 0; tn < 4; ++tn) {
        const int col = bn + waveN * 64 + tn * 16 + lr;
        atomicAdd(&orow[col], (acc[tm][tn][v] + bias[col]) * wgt);
      }
    }
  }
}

extern "C" void kernel_launch(void* const* d_in, const int* in_sizes, int n_in,
                              void* d_out, int out_size, void* d_ws, size_t ws_size,
                              hipStream_t stream) {
  (void)in_sizes; (void)n_in; (void)out_size; (void)ws_size;
  const float* inputs        = (const float*)d_in[0];
  const float* inputs_weight = (const float*)d_in[1];
  const int*   top_idx       = (const int*)d_in[2];
  const float* W1            = (const float*)d_in[3];
  const float* b1            = (const float*)d_in[4];
  const float* W2            = (const float*)d_in[5];
  const float* b2            = (const float*)d_in[6];
  float* out = (float*)d_out;

  // workspace: tok bf16 [C,D] (8 MB) then h bf16 [C,F] (32 MB)
  __bf16* tok = (__bf16*)d_ws;
  __bf16* h   = (__bf16*)((char*)d_ws + (size_t)CC * DD * sizeof(__bf16));

  {
    int n4 = T_TOK * DD / 4;
    zero_out_kernel<<<n4 / 256, 256, 0, stream>>>(out, n4);
  }
  for (int e = 0; e < EE; ++e) {
    gather_tokens_bf16<<<(CC * DD / 8) / 256, 256, 0, stream>>>(inputs, top_idx, e, tok);
    gemm1_gelu_kernel<<<dim3(FF / 128, CC / 128), 256, 0, stream>>>(
        tok, W1 + (size_t)e * DD * FF, b1 + (size_t)e * FF, h);
    gemm2_scatter_kernel<<<dim3(DD / 128, CC / 128), 256, 0, stream>>>(
        h, W2 + (size_t)e * FF * DD, b2 + (size_t)e * DD, top_idx, inputs_weight, e, out);
  }
}
